// SupportLayer_61718680043758
// MI455X (gfx1250) — matrix operand order repaired and checked by reading, not compiled, not run on hardware
//
#include <hip/hip_runtime.h>
#include <stdint.h>

// ---------------------------------------------------------------------------
// out[n,m] = sigmoid( relu(|e_n - s_m| W1 + b1) . W2 + b2 )
// N=256, M=1024, D=512, H=512  -> ~137 GFLOP, pure matrix-core compute.
// v_wmma_f32_16x16x32_bf16; A=|diff| built once in registers per wave;
// W1^T bf16 tiles double-buffered into LDS via global_load_async_to_lds_b128;
// B fragments software-pipelined through two register octets.
// ---------------------------------------------------------------------------

typedef __attribute__((ext_vector_type(16))) __bf16 v16bf;
typedef __attribute__((ext_vector_type(8)))  float  v8f;

#define NQ 256
#define MS 1024
#define DD 512
#define HH 512

#define WAVES   4
#define BLOCK   (WAVES * 32)
#define BSTRIDE 520   // padded bf16 row stride for LDS B tiles (bank spread)

static __device__ __forceinline__ unsigned short f2bf(float f) {
    union { float f; uint32_t u; } x; x.f = f;
    uint32_t u = x.u;
    u += 0x7FFFu + ((u >> 16) & 1u);   // round-to-nearest-even
    return (unsigned short)(u >> 16);
}

static __device__ __forceinline__ void diff8(const float4 a0, const float4 a1,
                                             const float4 b0, const float4 b1,
                                             unsigned short* d) {
    d[0] = f2bf(fabsf(a0.x - b0.x)); d[1] = f2bf(fabsf(a0.y - b0.y));
    d[2] = f2bf(fabsf(a0.z - b0.z)); d[3] = f2bf(fabsf(a0.w - b0.w));
    d[4] = f2bf(fabsf(a1.x - b1.x)); d[5] = f2bf(fabsf(a1.y - b1.y));
    d[6] = f2bf(fabsf(a1.z - b1.z)); d[7] = f2bf(fabsf(a1.w - b1.w));
}

// 8x async b128: stages 128 contiguous bytes of W1t into LDS, no VGPR bounce.
// Tracked by ASYNCcnt (CDNA5 ISA).
static __device__ __forceinline__ void issue8(const unsigned short* g, uint32_t lds) {
#pragma unroll
    for (int q = 0; q < 8; ++q) {
        asm volatile("global_load_async_to_lds_b128 %0, %1, off"
                     :: "v"(lds + q * 16), "v"(g + q * 8) : "memory");
    }
}
static __device__ __forceinline__ void wait_async8() {
    asm volatile("s_wait_asynccnt 0x8" ::: "memory");   // only newest batch may remain
}
static __device__ __forceinline__ void wait_async0() {
    asm volatile("s_wait_asynccnt 0x0" ::: "memory");
}

// ---- Prep: W1[D][H] fp32 (row-major) -> W1t[H][D] bf16 -------------------
__global__ void w1_transpose_bf16(const float* __restrict__ W1,
                                  unsigned short* __restrict__ W1t) {
    int idx = blockIdx.x * blockDim.x + threadIdx.x;
    if (idx >= DD * HH) return;
    int d = idx / HH, h = idx % HH;
    W1t[h * DD + d] = f2bf(W1[idx]);
}

// ---- Fused pairwise-MLP kernel -------------------------------------------
__launch_bounds__(BLOCK)
__global__ void support_fused(const float*          __restrict__ emb,
                              const float*          __restrict__ sup,
                              const unsigned short* __restrict__ W1t,
                              const float*          __restrict__ b1,
                              const float*          __restrict__ W2,
                              const float*          __restrict__ b2,
                              float*                __restrict__ out) {
    __shared__ unsigned short sB[2][16 * BSTRIDE];   // double-buffered W1t tile

    const int lane = threadIdx.x & 31;
    const int wave = threadIdx.x >> 5;
    const int mblocks = MS / 16;                          // 64
    const int n  = (blockIdx.x / mblocks) * WAVES + wave;
    const int m0 = (blockIdx.x % mblocks) * 16;

    const int row   = lane & 15;      // A row (support index), B/C column
    const int half  = lane >> 4;      // 0 | 1
    const int khalf = half * 8;       // A K-slice base within a 32-chunk

    // Cooperative staging map: 128 threads x 64 bf16 (128 B) each
    const int brow  = threadIdx.x >> 3;         // h column within tile
    const int bcol0 = (threadIdx.x & 7) * 64;   // K offset
    const unsigned short* gsrc0 = W1t + brow * DD + bcol0;
    const uint32_t lds0 = (uint32_t)(uintptr_t)&sB[0][brow * BSTRIDE + bcol0];
    const uint32_t lds1 = (uint32_t)(uintptr_t)&sB[1][brow * BSTRIDE + bcol0];

    // Kick off chunk 0 before the (VALU-heavy) A-fragment build.
    issue8(gsrc0, lds0);

    // ---- A fragments in registers: AF[kc] = 16x32 bf16 tile of |e-s|.
    // ISA 16-bit A layout: lanes<16 hold K{0..7,16..23}, lanes>=16 K{8..15,24..31}.
    union AU { v16bf v; unsigned short u[16]; };
    AU AF[16];
    const float4* e4 = (const float4*)(emb + n * DD);
    const float4* s4 = (const float4*)(sup + (m0 + row) * DD);
#pragma unroll
    for (int kc = 0; kc < 16; ++kc) {
        const int k0 = kc * 32 + khalf;
        const int k1 = k0 + 16;
        diff8(e4[k0 >> 2], e4[(k0 >> 2) + 1], s4[k0 >> 2], s4[(k0 >> 2) + 1], AF[kc].u);
        diff8(e4[k1 >> 2], e4[(k1 >> 2) + 1], s4[k1 >> 2], s4[(k1 >> 2) + 1], AF[kc].u + 8);
    }

    float partial[8];
#pragma unroll
    for (int i = 0; i < 8; ++i) partial[i] = 0.f;

    int cur = 0;
    for (int h0 = 0; h0 < HH; h0 += 16) {
        // All waves finished reading buf[cur^1] (previous iteration) here,
        // so it is safe to let new async writes land in it.
        __syncthreads();
        const bool pref = (h0 + 16) < HH;          // uniform
        if (pref) {
            issue8(gsrc0 + (h0 + 16) * DD, cur ? lds0 : lds1);
            wait_async8();                         // chunk h0's batch has landed
        } else {
            wait_async0();
        }
        __syncthreads();                           // every wave's quarter is in LDS

        // 4 independent accumulator chains to cover the WMMA RAW window
        // (ISA 7.12.1); B fragments software-pipelined through 2 octets so the
        // DS pipe runs one kc step ahead of the XDL pipe.
        v8f acc[4] = {{}, {}, {}, {}};
        const unsigned short* bbase = &sB[cur][row * BSTRIDE + half * 8];
        union BU { v16bf v; uint4 q[2]; };
        BU bf[2];
        {
            const uint4* bq = (const uint4*)(bbase);
            bf[0].q[0] = bq[0]; bf[0].q[1] = bq[2];
        }
#pragma unroll
        for (int kc = 0; kc < 16; ++kc) {
            if (kc < 15) {
                const uint4* bq = (const uint4*)(bbase + (kc + 1) * 32);
                bf[(kc + 1) & 1].q[0] = bq[0];
                bf[(kc + 1) & 1].q[1] = bq[2];
            }
            acc[kc & 3] = __builtin_amdgcn_wmma_f32_16x16x32_bf16(
                false, AF[kc].v, false, bf[kc & 1].v, (short)0, acc[kc & 3],
                false, false);
        }

        // h = relu(acc + b1); fold second einsum: partial += h * W2[col]
        const float b1v = b1[h0 + row];
        const float w2v = W2[h0 + row];
#pragma unroll
        for (int i = 0; i < 8; ++i) {
            float hv = (acc[0][i] + acc[1][i]) + (acc[2][i] + acc[3][i]) + b1v;
            hv = hv > 0.f ? hv : 0.f;
            partial[i] += hv * w2v;
        }
        cur ^= 1;
    }

    // C layout: VGPR i -> row (i + 8*half), col (lane&15). Reduce the 16
    // column-lanes within each half of the wave32.
#pragma unroll
    for (int i = 0; i < 8; ++i) {
#pragma unroll
        for (int off = 8; off >= 1; off >>= 1)
            partial[i] += __shfl_xor(partial[i], off, 16);
    }

    if ((lane & 15) == 0) {
        const float bias2 = b2[0];
        const int mbase = m0 + half * 8;
#pragma unroll
        for (int i = 0; i < 8; ++i) {
            float lg = partial[i] + bias2;
            out[n * MS + mbase + i] = 1.f / (1.f + __expf(-lg));
        }
    }
}

// ---------------------------------------------------------------------------
extern "C" void kernel_launch(void* const* d_in, const int* in_sizes, int n_in,
                              void* d_out, int out_size, void* d_ws, size_t ws_size,
                              hipStream_t stream) {
    (void)in_sizes; (void)n_in; (void)out_size; (void)ws_size;
    const float* emb = (const float*)d_in[0];
    const float* sup = (const float*)d_in[1];
    const float* W1  = (const float*)d_in[2];
    const float* b1  = (const float*)d_in[3];
    const float* W2  = (const float*)d_in[4];
    const float* b2  = (const float*)d_in[5];
    float* out = (float*)d_out;
    unsigned short* W1t = (unsigned short*)d_ws;   // 512*512 bf16 = 512 KB

    w1_transpose_bf16<<<(DD * HH + 255) / 256, 256, 0, stream>>>(W1, W1t);

    dim3 grid((NQ / WAVES) * (MS / 16));           // 4096 blocks x 128 threads
    support_fused<<<grid, BLOCK, 0, stream>>>(emb, sup, W1t, b1, W2, b2, out);
}
